// DDiTSparseBlock_45208825758305
// MI455X (gfx1250) — matrix operand order repaired and checked by reading, not compiled, not run on hardware
//
#include <hip/hip_runtime.h>
#include <hip/hip_bf16.h>

typedef __attribute__((ext_vector_type(16))) _Float16 v16h;
typedef __attribute__((ext_vector_type(8)))  float    v8f;
typedef __attribute__((ext_vector_type(4)))  unsigned int v4u;
typedef __attribute__((ext_vector_type(8)))  int v8i;
typedef __attribute__((ext_vector_type(4)))  int v4i;

#define DIM   512
#define NH    8
#define HD    64
#define P     64
#define NSEL  128
#define NNB   27
#define SEQ   4096
#define BATCH 4
#define MLPD  2048
#define QKVN  1536

#if __has_builtin(__builtin_amdgcn_tensor_load_to_lds) && __has_builtin(__builtin_amdgcn_s_wait_tensorcnt)
#define USE_TDM 1
#else
#define USE_TDM 0
#endif

// ---------------------------------------------------------------------------
// WMMA fragment helpers (contiguous k = 16*(lane/16) + i per lane, wave32)
// ---------------------------------------------------------------------------
__device__ __forceinline__ v8f wmma_f16(v16h a, v16h b, v8f c) {
  return __builtin_amdgcn_wmma_f32_16x16x32_f16(false, a, false, b, (short)0, c, false, false);
}

// A fragment 16x32 f16 from row-major src (pre-offset to (row0,k0)), leading dim ld.
__device__ __forceinline__ v16h ld_a(const _Float16* __restrict__ p, int ld, int lane) {
  int m = lane & 15, h = lane >> 4;
  const _Float16* q0 = p + (size_t)m * ld + 16 * h;
  v16h f;
#pragma unroll
  for (int i = 0; i < 16; ++i) f[i] = q0[i];
  return f;
}

// B fragment 32x16 where B[k][n] = src[n*ld + k]  (N x K row-major; contiguous per lane)
__device__ __forceinline__ v16h ld_b_nk(const _Float16* __restrict__ p, int ld, int lane) {
  int n = lane & 15, h = lane >> 4;
  const _Float16* q0 = p + (size_t)n * ld + 16 * h;
  v16h f;
#pragma unroll
  for (int i = 0; i < 16; ++i) f[i] = q0[i];
  return f;
}

// ---------------------------------------------------------------------------
// Tensor Data Mover: 2D tile (tile_elems x tile_rows, f16) global -> LDS
// ---------------------------------------------------------------------------
#if USE_TDM
typedef __attribute__((address_space(3))) const void* lds_cptr_t;
__device__ __forceinline__ unsigned lds_offset(const void* p) {
  return (unsigned)(unsigned long long)(lds_cptr_t)p;
}

__device__ __forceinline__ void tdm_load_2d(unsigned lds_off, const void* gptr,
                                            int tile_elems, int tile_rows, int row_stride) {
  unsigned long long ga = (unsigned long long)gptr;
  v4u g0;
  g0[0] = 1u;                                            // count=1, user mode
  g0[1] = lds_off;                                       // LDS byte address
  g0[2] = (unsigned)ga;                                  // global addr [31:0]
  g0[3] = (unsigned)((ga >> 32) & 0x01FFFFFFu) | (2u << 30);  // addr[56:32] | type=2
  v8i g1;
  g1[0] = (1 << 16);                                     // data_size = 2 bytes
  g1[1] = (tile_elems & 0xffff) << 16;                   // tensor_dim0 lo16
  g1[2] = (tile_rows & 0xffff) << 16;                    // tensor_dim0 hi | tensor_dim1 lo
  g1[3] = (tile_elems & 0xffff) << 16;                   // tensor_dim1 hi | tile_dim0
  g1[4] = (tile_rows & 0xffff);                          // tile_dim1 (tile_dim2=0)
  g1[5] = row_stride;                                    // tensor_dim0_stride lo32
  g1[6] = 0;
  g1[7] = 0;
  v4i z4 = {0, 0, 0, 0};
#if __has_include(<hip/amd_detail/amd_gfx1250_TDM.h>)
  v8i z8 = {0, 0, 0, 0, 0, 0, 0, 0};
  __builtin_amdgcn_tensor_load_to_lds(g0, g1, z4, z4, z8, 0);
#else
  __builtin_amdgcn_tensor_load_to_lds(g0, g1, z4, z4, 0);
#endif
}
#endif

// ---------------------------------------------------------------------------
// GEMM core: block = 8 waves x (16 rows x 64 cols); shared 64-col B panel
// staged in LDS in KC chunks, double-buffered via TDM.
// ---------------------------------------------------------------------------
#define KC 128
template <int K>
__device__ __forceinline__ void gemm_core_lds(const _Float16* __restrict__ a,     // A + row0*K
                                              const _Float16* __restrict__ wcol,  // W + col0*K
                                              _Float16* __restrict__ lds,         // 2*64*KC halves
                                              int tid, v8f acc[4]) {
  const int NCH = K / KC;
  int lane = tid & 31, wid = tid >> 5;
  v8f z = {0.f, 0.f, 0.f, 0.f, 0.f, 0.f, 0.f, 0.f};
#pragma unroll
  for (int j = 0; j < 4; ++j) acc[j] = z;
#if USE_TDM
  if (wid == 0) tdm_load_2d(lds_offset(lds), wcol, KC, 64, K);
#endif
  for (int c = 0; c < NCH; ++c) {
    _Float16* buf = lds + (c & 1) * (64 * KC);
#if USE_TDM
    if (wid == 0) __builtin_amdgcn_s_wait_tensorcnt(0);
    __syncthreads();                                   // buf[c] published
    if (wid == 0 && c + 1 < NCH)
      tdm_load_2d(lds_offset(lds + ((c + 1) & 1) * (64 * KC)),
                  wcol + (size_t)(c + 1) * KC, KC, 64, K);
#else
    {
      int r = tid >> 2, cc = (tid & 3) * 32;
      const _Float16* src = wcol + (size_t)r * K + c * KC + cc;
      _Float16* dstp = buf + r * KC + cc;
#pragma unroll
      for (int i = 0; i < 32; ++i) dstp[i] = src[i];
    }
    __syncthreads();
#endif
#pragma unroll
    for (int kk = 0; kk < KC; kk += 32) {
      v16h af = ld_a(a + c * KC + kk, K, lane);
      v16h b0 = ld_b_nk(buf + 0 * 16 * KC + kk, KC, lane);
      v16h b1 = ld_b_nk(buf + 1 * 16 * KC + kk, KC, lane);
      v16h b2 = ld_b_nk(buf + 2 * 16 * KC + kk, KC, lane);
      v16h b3 = ld_b_nk(buf + 3 * 16 * KC + kk, KC, lane);
      acc[0] = wmma_f16(af, b0, acc[0]);
      acc[1] = wmma_f16(af, b1, acc[1]);
      acc[2] = wmma_f16(af, b2, acc[2]);
      acc[3] = wmma_f16(af, b3, acc[3]);
    }
    __syncthreads();                                   // all done reading buf[c]
  }
}

// ---------------------------------------------------------------------------
// Small elementwise / setup kernels
// ---------------------------------------------------------------------------
__global__ void cvt_f16_kernel(const float* __restrict__ s, _Float16* __restrict__ d, int n) {
  int i = blockIdx.x * blockDim.x + threadIdx.x;
  if (i < n) d[i] = (_Float16)s[i];
}

__global__ void copy_f32_kernel(const float* __restrict__ s, float* __restrict__ d, int n) {
  int i = blockIdx.x * blockDim.x + threadIdx.x;
  if (i < n) d[i] = s[i];
}

__global__ void ada_kernel(const float* __restrict__ c, const float* __restrict__ w,
                           const float* __restrict__ b, float* __restrict__ ada) {
  int idx = blockIdx.x * blockDim.x + threadIdx.x;
  if (idx >= BATCH * 6 * DIM) return;
  int bb = idx / (6 * DIM), o = idx % (6 * DIM);
  const float* cv = c + bb * DIM;
  const float* wr = w + (size_t)o * DIM;
  float acc = b[o];
  for (int k = 0; k < DIM; ++k) acc += cv[k] * wr[k];
  ada[idx] = acc;
}

__global__ void selmap_kernel(const int* __restrict__ nz_b, const int* __restrict__ nz_s,
                              int* __restrict__ selof) {
  int t = threadIdx.x;
  if (t < 256) selof[t] = -1;
  __syncthreads();
  if (t < NSEL) selof[nz_b[t] * 64 + nz_s[t]] = t;
}

__device__ __forceinline__ int block_seq(int s, int p) {
  int i1 = s >> 4, i2 = (s >> 2) & 3, i3 = s & 3;
  int p1 = p >> 4, p2 = (p >> 2) & 3, p3 = p & 3;
  return p3 + 4 * i3 + 16 * p2 + 64 * i2 + 256 * p1 + 1024 * i1;
}

// ---------------------------------------------------------------------------
// LayerNorm + adaLN modulate
// ---------------------------------------------------------------------------
__global__ __launch_bounds__(256) void ln1_gather_kernel(
    const float* __restrict__ x, const int* __restrict__ nz_b, const int* __restrict__ nz_s,
    const float* __restrict__ ada, const float* __restrict__ nw, _Float16* __restrict__ dst) {
  __shared__ float red[256];
  int t = threadIdx.x;
  int blk = blockIdx.x;            // 0..8191 = n*64 + p
  int n = blk >> 6, p = blk & 63;
  int b = nz_b[n], s = nz_s[n];
  int seq = block_seq(s, p);
  const float* src = x + ((size_t)b * SEQ + seq) * DIM;
  float v0 = src[t], v1 = src[t + 256];
  red[t] = v0 + v1; __syncthreads();
  for (int st = 128; st > 0; st >>= 1) { if (t < st) red[t] += red[t + st]; __syncthreads(); }
  float mu = red[0] * (1.f / DIM); __syncthreads();
  float d0 = v0 - mu, d1 = v1 - mu;
  red[t] = d0 * d0 + d1 * d1; __syncthreads();
  for (int st = 128; st > 0; st >>= 1) { if (t < st) red[t] += red[t + st]; __syncthreads(); }
  float rstd = rsqrtf(red[0] * (1.f / DIM) + 1e-5f);
  const float* adab = ada + b * (6 * DIM);
  _Float16* dp = dst + (size_t)blk * DIM;
  dp[t]       = (_Float16)(d0 * rstd * nw[t]       * (1.f + adab[DIM + t])       + adab[t]);
  dp[t + 256] = (_Float16)(d1 * rstd * nw[t + 256] * (1.f + adab[DIM + t + 256]) + adab[t + 256]);
}

__global__ __launch_bounds__(256) void ln2_kernel(
    const float* __restrict__ xres, const float* __restrict__ ada,
    const float* __restrict__ nw, _Float16* __restrict__ dst) {
  __shared__ float red[256];
  int t = threadIdx.x;
  int row = blockIdx.x;            // 0..16383
  int b = row >> 12;
  const float* src = xres + (size_t)row * DIM;
  float v0 = src[t], v1 = src[t + 256];
  red[t] = v0 + v1; __syncthreads();
  for (int st = 128; st > 0; st >>= 1) { if (t < st) red[t] += red[t + st]; __syncthreads(); }
  float mu = red[0] * (1.f / DIM); __syncthreads();
  float d0 = v0 - mu, d1 = v1 - mu;
  red[t] = d0 * d0 + d1 * d1; __syncthreads();
  for (int st = 128; st > 0; st >>= 1) { if (t < st) red[t] += red[t + st]; __syncthreads(); }
  float rstd = rsqrtf(red[0] * (1.f / DIM) + 1e-5f);
  const float* adab = ada + b * (6 * DIM);
  _Float16* dp = dst + (size_t)row * DIM;
  dp[t]       = (_Float16)(d0 * rstd * nw[t]       * (1.f + adab[4 * DIM + t])       + adab[3 * DIM + t]);
  dp[t + 256] = (_Float16)(d1 * rstd * nw[t + 256] * (1.f + adab[4 * DIM + t + 256]) + adab[3 * DIM + t + 256]);
}

// ---------------------------------------------------------------------------
// GEMM kernels
// ---------------------------------------------------------------------------
__global__ __launch_bounds__(256) void gemm_qkv_kernel(
    const _Float16* __restrict__ A, const _Float16* __restrict__ W, float* __restrict__ C) {
  __shared__ _Float16 Bsh[2 * 64 * KC];
  int tid = threadIdx.x, lane = tid & 31, wid = tid >> 5;
  int row0 = blockIdx.x * 128 + wid * 16, col0 = blockIdx.y * 64;
  v8f acc[4];
  gemm_core_lds<DIM>(A + (size_t)row0 * DIM, W + (size_t)col0 * DIM, Bsh, tid, acc);
  int h = lane >> 4, nl = lane & 15;
#pragma unroll
  for (int j = 0; j < 4; ++j)
#pragma unroll
    for (int r = 0; r < 8; ++r)
      C[(size_t)(row0 + 8 * h + r) * QKVN + col0 + j * 16 + nl] = acc[j][r];
}

__global__ __launch_bounds__(256) void gemm_outproj_kernel(
    const _Float16* __restrict__ A, const _Float16* __restrict__ W, float* __restrict__ C) {
  __shared__ _Float16 Bsh[2 * 64 * KC];
  int tid = threadIdx.x, lane = tid & 31, wid = tid >> 5;
  int row0 = blockIdx.x * 128 + wid * 16, col0 = blockIdx.y * 64;
  v8f acc[4];
  gemm_core_lds<DIM>(A + (size_t)row0 * DIM, W + (size_t)col0 * DIM, Bsh, tid, acc);
  int h = lane >> 4, nl = lane & 15;
#pragma unroll
  for (int j = 0; j < 4; ++j)
#pragma unroll
    for (int r = 0; r < 8; ++r)
      C[(size_t)(row0 + 8 * h + r) * DIM + col0 + j * 16 + nl] = acc[j][r];
}

__global__ __launch_bounds__(256) void gemm_mlp1_kernel(
    const _Float16* __restrict__ A, const _Float16* __restrict__ W,
    const float* __restrict__ bias, _Float16* __restrict__ C) {
  __shared__ _Float16 Bsh[2 * 64 * KC];
  int tid = threadIdx.x, lane = tid & 31, wid = tid >> 5;
  int row0 = blockIdx.x * 128 + wid * 16, col0 = blockIdx.y * 64;
  v8f acc[4];
  gemm_core_lds<DIM>(A + (size_t)row0 * DIM, W + (size_t)col0 * DIM, Bsh, tid, acc);
  int h = lane >> 4, nl = lane & 15;
#pragma unroll
  for (int j = 0; j < 4; ++j)
#pragma unroll
    for (int r = 0; r < 8; ++r) {
      int col = col0 + j * 16 + nl;
      float v = acc[j][r] + bias[col];
      float g = 0.5f * v * (1.f + tanhf(0.7978845608028654f * (v + 0.044715f * v * v * v)));
      C[(size_t)(row0 + 8 * h + r) * MLPD + col] = (_Float16)g;
    }
}

__global__ __launch_bounds__(256) void gemm_mlp2_kernel(
    const _Float16* __restrict__ A, const _Float16* __restrict__ W,
    const float* __restrict__ bias, const float* __restrict__ xres,
    const float* __restrict__ ada, float* __restrict__ out) {
  __shared__ _Float16 Bsh[2 * 64 * KC];
  int tid = threadIdx.x, lane = tid & 31, wid = tid >> 5;
  int row0 = blockIdx.x * 128 + wid * 16, col0 = blockIdx.y * 64;
  v8f acc[4];
  gemm_core_lds<MLPD>(A + (size_t)row0 * MLPD, W + (size_t)col0 * MLPD, Bsh, tid, acc);
  int h = lane >> 4, nl = lane & 15;
#pragma unroll
  for (int j = 0; j < 4; ++j)
#pragma unroll
    for (int r = 0; r < 8; ++r) {
      int row = row0 + 8 * h + r, col = col0 + j * 16 + nl;
      int b = row >> 12;
      float g = ada[b * (6 * DIM) + 5 * DIM + col];
      size_t o = (size_t)row * DIM + col;
      out[o] = xres[o] + g * (acc[j][r] + bias[col]);
    }
}

// ---------------------------------------------------------------------------
// RoPE: qkv f32 (8192 x 1536) -> q_rope/k_rope f16 (8192 x 512, row-major) and
// V pre-TRANSPOSED per (block, head): vT[m][h][d][p] (contiguous 64x64 tiles)
// so the attention PV B-fragments are lane-contiguous LDS reads.
// ---------------------------------------------------------------------------
__global__ void rope_kernel(const float* __restrict__ qkv,
                            const float* __restrict__ cq, const float* __restrict__ sq,
                            const float* __restrict__ ck, const float* __restrict__ sk,
                            _Float16* __restrict__ qo, _Float16* __restrict__ ko,
                            _Float16* __restrict__ vT) {
  int idx = blockIdx.x * blockDim.x + threadIdx.x;
  if (idx >= NSEL * P * DIM) return;
  int row = idx >> 9, col = idx & 511;
  int p = row & 63, d = col & 63;
  int colr = (d < 32) ? col + 32 : col - 32;
  float sgn = (d < 32) ? -1.f : 1.f;
  const float* base = qkv + (size_t)row * QKVN;
  float qv = base[col], kv = base[DIM + col], vv = base[2 * DIM + col];
  float qr = sgn * base[colr], kr = sgn * base[DIM + colr];
  qo[idx] = (_Float16)(qv * cq[p * HD + d] + qr * sq[p * HD + d]);
  ko[idx] = (_Float16)(kv * ck[p * HD + d] + kr * sk[p * HD + d]);
  int m = row >> 6, hh = col >> 6;
  vT[(size_t)(m * NH + hh) * 4096 + d * 64 + p] = (_Float16)vv;
}

__global__ void empty_rope_kernel(const float* __restrict__ kv_emp,
                                  const float* __restrict__ ck, const float* __restrict__ sk,
                                  _Float16* __restrict__ ke, _Float16* __restrict__ veT) {
  int idx = blockIdx.x * blockDim.x + threadIdx.x;   // 64*512
  if (idx >= P * DIM) return;
  int p = idx >> 9, col = idx & 511, d = col & 63, hh = col >> 6;
  int colr = (d < 32) ? col + 32 : col - 32;
  float sgn = (d < 32) ? -1.f : 1.f;
  const float* base = kv_emp + (size_t)p * (2 * DIM);
  float kv = base[col], kr = sgn * base[colr];
  ke[idx] = (_Float16)(kv * ck[p * HD + d] + kr * sk[p * HD + d]);
  veT[(size_t)hh * 4096 + d * 64 + p] = (_Float16)base[DIM + col];
}

// ---------------------------------------------------------------------------
// Flash attention: one workgroup per (selected block, head); 4 waves x 16 rows.
// K tile [key][d] (stride DIM in global), V tile pre-transposed [d][key]
// (contiguous 4096 halves). Both double-buffered in LDS via TDM.
// ---------------------------------------------------------------------------
__global__ __launch_bounds__(128) void attn_kernel(
    const _Float16* __restrict__ qh, const _Float16* __restrict__ kh,
    const _Float16* __restrict__ vT, const _Float16* __restrict__ k_emp,
    const _Float16* __restrict__ veT, const int* __restrict__ nb_idx,
    const int* __restrict__ selof, _Float16* __restrict__ outh) {
  __shared__ _Float16 Kb[2][64 * 64];
  __shared__ _Float16 Vb[2][64 * 64];
  __shared__ _Float16 Ss[4][16 * 64];
  int n = blockIdx.x >> 3, hd = blockIdx.x & 7;
  int wid = threadIdx.x >> 5, lane = threadIdx.x & 31;
  int half = lane >> 4, nl = lane & 15;

  const _Float16* qbase = qh + ((size_t)(n * 64 + wid * 16)) * DIM + hd * HD;
  v16h aq0 = ld_a(qbase, DIM, lane);
  v16h aq1 = ld_a(qbase + 32, DIM, lane);

  v8f z = {0.f, 0.f, 0.f, 0.f, 0.f, 0.f, 0.f, 0.f};
  v8f o[4]; float mrow[8], lrow[8];
#pragma unroll
  for (int j = 0; j < 4; ++j) o[j] = z;
#pragma unroll
  for (int r = 0; r < 8; ++r) { mrow[r] = -1e30f; lrow[r] = 0.f; }

#define KV_BASE(jj, kb, vb)                                                       \
  {                                                                               \
    int g_ = nb_idx[n * NNB + (jj)];                                              \
    int m_ = selof[g_];                                                           \
    kb = (m_ >= 0) ? kh + (size_t)m_ * P * DIM + hd * HD : k_emp + hd * HD;       \
    vb = (m_ >= 0) ? vT + (size_t)(m_ * NH + hd) * 4096 : veT + (size_t)hd * 4096;\
  }

#if USE_TDM
  if (wid == 0) {
    const _Float16 *kb, *vb;
    KV_BASE(0, kb, vb);
    tdm_load_2d(lds_offset(&Kb[0][0]), kb, 64, 64, DIM);
    tdm_load_2d(lds_offset(&Vb[0][0]), vb, 64, 64, 64);
  }
#endif

  for (int j = 0; j < NNB; ++j) {
    int cur = j & 1;
#if USE_TDM
    if (wid == 0) __builtin_amdgcn_s_wait_tensorcnt(0);
    __syncthreads();                         // tiles for j published
    if (wid == 0 && j + 1 < NNB) {
      const _Float16 *kb, *vb;
      KV_BASE(j + 1, kb, vb);
      tdm_load_2d(lds_offset(&Kb[(j + 1) & 1][0]), kb, 64, 64, DIM);
      tdm_load_2d(lds_offset(&Vb[(j + 1) & 1][0]), vb, 64, 64, 64);
    }
#else
    {
      const _Float16 *kb, *vb;
      KV_BASE(j, kb, vb);
      int r = threadIdx.x >> 1, cc = (threadIdx.x & 1) * 32;
#pragma unroll
      for (int i = 0; i < 32; ++i) {
        Kb[cur][r * 64 + cc + i] = kb[(size_t)r * DIM + cc + i];
        Vb[cur][r * 64 + cc + i] = vb[(size_t)r * 64 + cc + i];
      }
    }
    __syncthreads();
#endif
    const _Float16* KT = Kb[cur];
    const _Float16* VTl = Vb[cur];

    // S = Q K^T (16 x 64 per wave); B[k=d][n=key] = KT[key*64 + d]
    v8f s[4];
#pragma unroll
    for (int c = 0; c < 4; ++c) {
      s[c] = z;
      v16h b0 = ld_b_nk(KT + (size_t)(c * 16) * 64, 64, lane);
      s[c] = wmma_f16(aq0, b0, s[c]);
      v16h b1 = ld_b_nk(KT + (size_t)(c * 16) * 64 + 32, 64, lane);
      s[c] = wmma_f16(aq1, b1, s[c]);
    }
    // online softmax (rows live in 16-lane groups of the C layout)
    float alpha[8];
#pragma unroll
    for (int r = 0; r < 8; ++r) {
      float mx = -1e30f;
#pragma unroll
      for (int c = 0; c < 4; ++c) mx = fmaxf(mx, s[c][r] * 0.125f);
      for (int d = 1; d < 16; d <<= 1) mx = fmaxf(mx, __shfl_xor(mx, d));
      float mnew = fmaxf(mrow[r], mx);
      alpha[r] = __expf(mrow[r] - mnew);
      mrow[r] = mnew;
    }
#pragma unroll
    for (int r = 0; r < 8; ++r) {
      float sum = 0.f;
#pragma unroll
      for (int c = 0; c < 4; ++c) {
        float pv = __expf(s[c][r] * 0.125f - mrow[r]);
        s[c][r] = pv;
        sum += pv;
      }
      for (int d = 1; d < 16; d <<= 1) sum += __shfl_xor(sum, d);
      lrow[r] = lrow[r] * alpha[r] + sum;
#pragma unroll
      for (int c = 0; c < 4; ++c) o[c][r] *= alpha[r];
    }
    // reshape P (C-layout) -> A-fragments via per-wave LDS scratch
    _Float16* ss = Ss[wid];
#pragma unroll
    for (int c = 0; c < 4; ++c)
#pragma unroll
      for (int r = 0; r < 8; ++r)
        ss[(r + 8 * half) * 64 + c * 16 + nl] = (_Float16)s[c][r];
    v16h ap0 = ld_a(ss, 64, lane);
    v16h ap1 = ld_a(ss + 32, 64, lane);
    // O += P V; V transposed in LDS: B[k=key][n=d] = VTl[d*64 + key] -> ld_b_nk
#pragma unroll
    for (int c = 0; c < 4; ++c) {
      v16h bv0 = ld_b_nk(VTl + (size_t)(c * 16) * 64, 64, lane);
      o[c] = wmma_f16(ap0, bv0, o[c]);
      v16h bv1 = ld_b_nk(VTl + (size_t)(c * 16) * 64 + 32, 64, lane);
      o[c] = wmma_f16(ap1, bv1, o[c]);
    }
    __syncthreads();                         // all waves done with tiles j
  }
#pragma unroll
  for (int c = 0; c < 4; ++c)
#pragma unroll
    for (int r = 0; r < 8; ++r) {
      int row = n * 64 + wid * 16 + 8 * half + r;
      int col = hd * HD + c * 16 + nl;
      outh[(size_t)row * DIM + col] = (_Float16)(o[c][r] / lrow[r]);
    }
}

// ---------------------------------------------------------------------------
// scatter: xres[sel tokens] += g_msa * y
// ---------------------------------------------------------------------------
__global__ void scatter_res_kernel(const float* __restrict__ y, const int* __restrict__ nz_b,
                                   const int* __restrict__ nz_s, const float* __restrict__ ada,
                                   float* __restrict__ xres) {
  int idx = blockIdx.x * blockDim.x + threadIdx.x;
  if (idx >= NSEL * P * DIM) return;
  int row = idx >> 9, d = idx & 511;
  int n = row >> 6, p = row & 63;
  int b = nz_b[n], s = nz_s[n];
  int seq = block_seq(s, p);
  size_t a = ((size_t)b * SEQ + seq) * DIM + d;
  float g = ada[b * (6 * DIM) + 2 * DIM + d];
  xres[a] += g * y[idx];
}

// ---------------------------------------------------------------------------
extern "C" void kernel_launch(void* const* d_in, const int* in_sizes, int n_in,
                              void* d_out, int out_size, void* d_ws, size_t ws_size,
                              hipStream_t stream) {
  const float* x      = (const float*)d_in[0];
  const float* c      = (const float*)d_in[1];
  const float* cosq   = (const float*)d_in[2];
  const float* sinq   = (const float*)d_in[3];
  const float* cosk   = (const float*)d_in[4];
  const float* sink   = (const float*)d_in[5];
  const int*   nz_b   = (const int*)d_in[6];
  const int*   nz_s   = (const int*)d_in[7];
  const int*   nbidx  = (const int*)d_in[8];
  const float* n1w    = (const float*)d_in[9];
  const float* n2w    = (const float*)d_in[10];
  const float* w_qkv  = (const float*)d_in[11];
  const float* w_out  = (const float*)d_in[12];
  const float* w_mlp1 = (const float*)d_in[13];
  const float* b_mlp1 = (const float*)d_in[14];
  const float* w_mlp2 = (const float*)d_in[15];
  const float* b_mlp2 = (const float*)d_in[16];
  const float* w_ada  = (const float*)d_in[17];
  const float* b_ada  = (const float*)d_in[18];
  const float* kv_emp = (const float*)d_in[19];
  float* out = (float*)d_out;

  char* ws = (char*)d_ws;
  size_t off = 0;
  auto alloc = [&](size_t bytes) -> char* {
    char* p = ws + off;
    off += (bytes + 255) & ~(size_t)255;
    return p;
  };
  float*    ada     = (float*)   alloc((size_t)BATCH * 6 * DIM * 4);
  int*      selof   = (int*)     alloc(256 * 4);
  _Float16* wqkv_h  = (_Float16*)alloc((size_t)QKVN * DIM * 2);
  _Float16* wout_h  = (_Float16*)alloc((size_t)DIM * DIM * 2);
  _Float16* wmlp1_h = (_Float16*)alloc((size_t)MLPD * DIM * 2);
  _Float16* wmlp2_h = (_Float16*)alloc((size_t)DIM * MLPD * 2);
  _Float16* kemp_h  = (_Float16*)alloc((size_t)P * DIM * 2);
  _Float16* vempT_h = (_Float16*)alloc((size_t)P * DIM * 2);
  _Float16* xsel_h  = (_Float16*)alloc((size_t)NSEL * P * DIM * 2);
  _Float16* q_h     = (_Float16*)alloc((size_t)NSEL * P * DIM * 2);
  _Float16* k_h     = (_Float16*)alloc((size_t)NSEL * P * DIM * 2);
  _Float16* vT_h    = (_Float16*)alloc((size_t)NSEL * P * DIM * 2);
  _Float16* attn_h  = (_Float16*)alloc((size_t)NSEL * P * DIM * 2);
  float*    xres    = (float*)   alloc((size_t)BATCH * SEQ * DIM * 4);
  // Region R: h2_h | (m1_h aliased with qkv_f32 and y_proj — disjoint lifetimes)
  char* R = alloc((size_t)16777216 + 67108864);
  _Float16* h2_h   = (_Float16*)R;               // 16384x512 f16 (stage 7+)
  float*    qkv_f  = (float*)   (R + 16777216);  // 8192x1536 f32 (stages 2-3)
  float*    y_proj = (float*)   (R + 16777216);  // 8192x512 f32 (stage 5)
  _Float16* m1_h   = (_Float16*)(R + 16777216);  // 16384x2048 f16 (stages 7-8)

  const int T = 256;
  hipLaunchKernelGGL(ada_kernel, dim3((BATCH * 6 * DIM + T - 1) / T), dim3(T), 0, stream,
                     c, w_ada, b_ada, ada);
  hipLaunchKernelGGL(selmap_kernel, dim3(1), dim3(256), 0, stream, nz_b, nz_s, selof);
  hipLaunchKernelGGL(cvt_f16_kernel, dim3((QKVN * DIM + T - 1) / T), dim3(T), 0, stream,
                     w_qkv, wqkv_h, QKVN * DIM);
  hipLaunchKernelGGL(cvt_f16_kernel, dim3((DIM * DIM + T - 1) / T), dim3(T), 0, stream,
                     w_out, wout_h, DIM * DIM);
  hipLaunchKernelGGL(cvt_f16_kernel, dim3((MLPD * DIM + T - 1) / T), dim3(T), 0, stream,
                     w_mlp1, wmlp1_h, MLPD * DIM);
  hipLaunchKernelGGL(cvt_f16_kernel, dim3((DIM * MLPD + T - 1) / T), dim3(T), 0, stream,
                     w_mlp2, wmlp2_h, DIM * MLPD);
  hipLaunchKernelGGL(empty_rope_kernel, dim3((P * DIM + T - 1) / T), dim3(T), 0, stream,
                     kv_emp, cosk, sink, kemp_h, vempT_h);
  hipLaunchKernelGGL(ln1_gather_kernel, dim3(NSEL * P), dim3(T), 0, stream,
                     x, nz_b, nz_s, ada, n1w, xsel_h);
  hipLaunchKernelGGL(gemm_qkv_kernel, dim3(NSEL * P / 128, QKVN / 64), dim3(T), 0, stream,
                     xsel_h, wqkv_h, qkv_f);
  hipLaunchKernelGGL(rope_kernel, dim3((NSEL * P * DIM + T - 1) / T), dim3(T), 0, stream,
                     qkv_f, cosq, sinq, cosk, sink, q_h, k_h, vT_h);
  hipLaunchKernelGGL(attn_kernel, dim3(NSEL * NH), dim3(128), 0, stream,
                     q_h, k_h, vT_h, kemp_h, vempT_h, nbidx, selof, attn_h);
  hipLaunchKernelGGL(gemm_outproj_kernel, dim3(NSEL * P / 128, DIM / 64), dim3(T), 0, stream,
                     attn_h, wout_h, y_proj);
  hipLaunchKernelGGL(copy_f32_kernel, dim3((BATCH * SEQ * DIM + T - 1) / T), dim3(T), 0, stream,
                     x, xres, BATCH * SEQ * DIM);
  hipLaunchKernelGGL(scatter_res_kernel, dim3((NSEL * P * DIM + T - 1) / T), dim3(T), 0, stream,
                     y_proj, nz_b, nz_s, ada, xres);
  hipLaunchKernelGGL(ln2_kernel, dim3(BATCH * SEQ), dim3(T), 0, stream, xres, ada, n2w, h2_h);
  hipLaunchKernelGGL(gemm_mlp1_kernel, dim3(BATCH * SEQ / 128, MLPD / 64), dim3(T), 0, stream,
                     h2_h, wmlp1_h, b_mlp1, m1_h);
  hipLaunchKernelGGL(gemm_mlp2_kernel, dim3(BATCH * SEQ / 128, DIM / 64), dim3(T), 0, stream,
                     m1_h, wmlp2_h, b_mlp2, xres, ada, out);
}
